// STGCNBlock_28329604284583
// MI455X (gfx1250) — compile-verified, matrix operand order checked by reading, not run on hardware
//
#include <hip/hip_runtime.h>
#include <hip/hip_bf16.h>

typedef __attribute__((ext_vector_type(16))) _Float16 v16h;
typedef __attribute__((ext_vector_type(8)))  _Float16 v8h;
typedef __attribute__((ext_vector_type(8)))  float    v8f;
typedef unsigned int u32x4 __attribute__((ext_vector_type(4)));
typedef int          i32x8 __attribute__((ext_vector_type(8)));
typedef int          i32x4 __attribute__((ext_vector_type(4)));

#define B_      32
#define CIN     64
#define COUT    128
#define T_      300
#define V_      25
#define NCOL    (T_ * V_)            // 7500
#define NTILES  ((NCOL + 15) / 16)   // 469
#define NX      ((size_t)B_ * CIN * NCOL)   // 15,360,000
#define NY      ((size_t)B_ * COUT * NCOL)  // 30,720,000
#define BN_EPS  1e-5f
#define HALO    100                  // 4 taps * 25 columns
#define SLABW   216                  // 16 + 2*HALO
#define TT      64                   // transpose tile

#if defined(__HIP_DEVICE_COMPILE__) && __has_builtin(__builtin_amdgcn_tensor_load_to_lds)
#define HAVE_TDM 1
#else
#define HAVE_TDM 0
#endif

// Compiler-level fence: the TDM engine wrote LDS behind the compiler's back.
// Escape the slab pointer into an opaque asm with a memory clobber so slab
// reads cannot be folded to undef and must become real ds_load instructions.
__device__ __forceinline__ void lds_written_by_dma(const void* p) {
  asm volatile("" : : "v"(p) : "memory");
}

// ---------------------------------------------------------------------------
// TDM: 2-D f16 tile load Global -> LDS via tensor descriptor (ISA ch.8).
// Elements beyond tensor_w are zero-filled by the DMA (OOB-read-returns-zero),
// which we use to implement zero padding with a constant LDS row stride.
// ---------------------------------------------------------------------------
#if HAVE_TDM
__device__ __forceinline__ void tdm_load_2d_f16(unsigned lds_off, const void* gaddr,
                                                unsigned tile_w, unsigned tile_h,
                                                unsigned tensor_w, unsigned tensor_h,
                                                unsigned row_stride_elems) {
  unsigned long long ga = (unsigned long long)gaddr;
  u32x4 g0;
  g0[0] = 1u;                                              // count=1, no gather/restore
  g0[1] = lds_off;                                         // lds_addr [63:32]
  g0[2] = (unsigned)(ga & 0xFFFFFFFFu);                    // global_addr low
  g0[3] = (unsigned)((ga >> 32) & 0x01FFFFFFu) | (2u << 30); // addr[56:32] | type=2
  i32x8 g1;
  g1[0] = (int)(1u << 16);                                 // data_size=1 (2 bytes)
  g1[1] = (int)((tensor_w & 0xFFFFu) << 16);               // tensor_dim0[15:0] -> bits 63:48
  g1[2] = (int)(((tensor_w >> 16) & 0xFFFFu) | ((tensor_h & 0xFFFFu) << 16));
  g1[3] = (int)(((tensor_h >> 16) & 0xFFFFu) | ((tile_w & 0xFFFFu) << 16));
  g1[4] = (int)(tile_h & 0xFFFFu);                         // tile_dim1 (tile_dim2=0)
  g1[5] = (int)row_stride_elems;                           // tensor_dim0_stride low32
  g1[6] = 0;                                               // stride hi16 | dim1_stride lo16
  g1[7] = 0;
  i32x4 z4 = {0, 0, 0, 0};
#if __clang_major__ >= 23
  i32x8 z8 = {0, 0, 0, 0, 0, 0, 0, 0};
  __builtin_amdgcn_tensor_load_to_lds(g0, g1, z4, z4, z8, 0);
#else
  __builtin_amdgcn_tensor_load_to_lds(g0, g1, z4, z4, 0);
#endif
}
#endif

// build v16h operand from two contiguous 8-half runs (K and K+16); works for
// global (b128 loads) and LDS (ds_load_b128) pointers alike.
__device__ __forceinline__ v16h load_a16(const _Float16* p) {
  const v8h* vp = (const v8h*)p;
  v8h lo = vp[0], hi = vp[2];   // +16 halves
  return __builtin_shufflevector(lo, hi, 0, 1, 2, 3, 4, 5, 6, 7,
                                 8, 9, 10, 11, 12, 13, 14, 15);
}

// ---------------------------------------------------------------------------
// Elementwise helpers
// ---------------------------------------------------------------------------
__global__ void cvt_f32_to_f16(const float* __restrict__ in, _Float16* __restrict__ out, size_t n) {
  size_t i = (size_t)blockIdx.x * blockDim.x + threadIdx.x;
  if (i < n) out[i] = (_Float16)in[i];
}

// Reorder w_t (O=128, C=128, K=9, 1) f32 -> (K, O, C) f16
__global__ void reorder_wt(const float* __restrict__ wt, _Float16* __restrict__ out) {
  int i = blockIdx.x * blockDim.x + threadIdx.x;
  if (i >= COUT * COUT * 9) return;
  int k = i / (COUT * COUT);
  int rem = i % (COUT * COUT);
  int o = rem / COUT, c = rem % COUT;
  out[i] = (_Float16)wt[(size_t)o * (COUT * 9) + (size_t)c * 9 + k];
}

__global__ void zero_f32(float* __restrict__ p, int n) {
  int i = blockIdx.x * blockDim.x + threadIdx.x;
  if (i < n) p[i] = 0.f;
}

// ---------------------------------------------------------------------------
// Graph mixing: xg[b,c,t,v] = sum_w A[v,w] * x[b,c,t,w]  (output f16)
// ---------------------------------------------------------------------------
__global__ void graph_mix(const float* __restrict__ x, const float* __restrict__ A,
                          _Float16* __restrict__ xg, size_t total) {
  __shared__ float As[V_ * V_];
  for (int i = threadIdx.x; i < V_ * V_; i += blockDim.x) As[i] = A[i];
  __syncthreads();
  size_t idx = (size_t)blockIdx.x * blockDim.x + threadIdx.x;
  if (idx >= total) return;
  size_t r = idx / V_;
  int v = (int)(idx % V_);
  const float* xr = x + r * V_;
  float s = 0.f;
#pragma unroll
  for (int w = 0; w < V_; ++w) s += As[v * V_ + w] * xr[w];
  xg[idx] = (_Float16)s;
}

// ---------------------------------------------------------------------------
// Pointwise GEMM: Y_b(128x7500) = W(128x64) @ X_b(64x7500) + bias.
// X tile (64x16) staged to LDS by TDM with tile_w=16 always (DMA zero-fills
// OOB columns) -> constant LDS stride, unconditional slab reads.
// ---------------------------------------------------------------------------
__global__ void __launch_bounds__(256)
gemm_pointwise_f16(const _Float16* __restrict__ X,   // (B, 64, 7500)
                   const _Float16* __restrict__ W,   // (128, 64) row-major
                   const float* __restrict__ bias,   // (128)
                   float* __restrict__ Y) {          // (B, 128, 7500)
  __shared__ _Float16 slab[CIN * 16];                // 2 KB, row stride 16
  int tile = blockIdx.x;
  int bb = tile / NTILES, ntile = tile % NTILES;
  int n0 = ntile * 16;
  const _Float16* Xb = X + (size_t)bb * CIN * NCOL;

#if HAVE_TDM
  if (threadIdx.x < 32) {
    tdm_load_2d_f16(0u, (const void*)(Xb + n0), 16u, CIN,
                    (unsigned)(NCOL - n0), CIN, NCOL);
    __builtin_amdgcn_s_wait_tensorcnt(0);
  }
#else
  for (int idx = threadIdx.x; idx < CIN * 16; idx += 256) {
    int r = idx >> 4, cc = idx & 15;
    slab[idx] = (n0 + cc < NCOL) ? Xb[(size_t)r * NCOL + n0 + cc] : (_Float16)0.f;
  }
#endif
  __syncthreads();
  lds_written_by_dma(slab);          // slab now holds DMA-written data

  int wave = threadIdx.x >> 5, lane = threadIdx.x & 31;
  int m0 = wave * 16;
  int lm = lane & 15;
  int lh = (lane >> 4) & 1;
  int n = n0 + lm;

  const _Float16* Wrow = W + (size_t)(m0 + lm) * CIN + (lh ? 8 : 0);
  v16h a0 = load_a16(Wrow);          // K 0..31 slice for this half-wave
  v16h a1 = load_a16(Wrow + 32);     // K 32..63

  v16h bv0, bv1;
#pragma unroll
  for (int i = 0; i < 16; ++i) {
    int koff = (i & 7) + ((i & 8) ? 16 : 0) + (lh ? 8 : 0);
    bv0[i] = slab[koff * 16 + lm];          // zero-filled beyond NCOL
    bv1[i] = slab[(32 + koff) * 16 + lm];
  }
  v8f acc = {};
  acc = __builtin_amdgcn_wmma_f32_16x16x32_f16(false, a0, false, bv0, (short)0, acc, false, false);
  acc = __builtin_amdgcn_wmma_f32_16x16x32_f16(false, a1, false, bv1, (short)0, acc, false, false);

  if (n < NCOL) {
    float* Yb = Y + (size_t)bb * COUT * NCOL;
#pragma unroll
    for (int r = 0; r < 8; ++r) {
      int m = m0 + (lh ? 8 : 0) + r;   // D layout: lanes>=16 hold M=8..15
      Yb[(size_t)m * NCOL + n] = acc[r] + bias[m];
    }
  }
}

// ---------------------------------------------------------------------------
// BN + ReLU + f16 + transpose: h1 (B, COUT, NCOL) f32 -> h1t (B, NCOL, COUT)
// f16 via 64x64 LDS tile (padded stride 65). Coalesced reads AND writes; this
// produces the layout where the temporal-conv B operand is contiguous per lane.
// ---------------------------------------------------------------------------
__global__ void __launch_bounds__(256)
norm_relu_transpose_f16(const float* __restrict__ Y,      // (B, COUT, NCOL)
                        const float* __restrict__ stats,
                        const float* __restrict__ g, const float* __restrict__ be,
                        _Float16* __restrict__ out) {     // (B, NCOL, COUT)
  __shared__ _Float16 tile[TT][TT + 1];                   // 8,320 B
  const int CTILES = COUT / TT;                           // 2
  const int COLT = (NCOL + TT - 1) / TT;                  // 118
  int blk = blockIdx.x;
  int ct = blk % CTILES;
  int colt = (blk / CTILES) % COLT;
  int b = blk / (CTILES * COLT);
  int c0 = ct * TT, col0 = colt * TT;
  const float* Yb = Y + (size_t)b * COUT * NCOL;

  for (int idx = threadIdx.x; idx < TT * TT; idx += 256) {
    int r = idx >> 6, cc = idx & 63;       // r: c offset, cc: col offset
    int c = c0 + r, col = col0 + cc;
    float v = 0.f;
    if (col < NCOL) {
      v = (Yb[(size_t)c * NCOL + col] - stats[2 * COUT + c]) * stats[3 * COUT + c] * g[c] + be[c];
      v = fmaxf(v, 0.f);
    }
    tile[r][cc] = (_Float16)v;
  }
  __syncthreads();

  _Float16* ob = out + (size_t)b * NCOL * COUT;
  for (int idx = threadIdx.x; idx < TT * TT; idx += 256) {
    int rr = idx >> 6, cc = idx & 63;      // rr: col offset, cc: c offset
    int col = col0 + rr;
    if (col < NCOL) ob[(size_t)col * COUT + c0 + cc] = tile[cc][rr];
  }
}

// ---------------------------------------------------------------------------
// Temporal conv 9x1 (pad 4) on transposed input Ht (B, NCOL, COUT):
// the slab is [col][c] (216x128 f16, 55KB) -- a fully CONTIGUOUS global chunk,
// staged as a 1-D TDM copy (right-edge zeros via tensor_w; left-edge blocks,
// 7 of 469, use a cooperative zero-padded copy). Each WMMA B-operand is then
// two contiguous 8-half runs -> 2x ds_load_b128. 36 WMMAs per 16x16 tile.
// ---------------------------------------------------------------------------
__global__ void __launch_bounds__(256)
temporal_conv_wmma(const _Float16* __restrict__ Ht,  // (B, NCOL, COUT) f16
                   const _Float16* __restrict__ Wt,  // (9, 128, 128) [k][o][c]
                   const float* __restrict__ bias,   // (128)
                   float* __restrict__ Y) {          // (B, COUT, NCOL)
  __shared__ _Float16 slab[SLABW * COUT];            // 55,296 B, [col][c]
  int tile = blockIdx.x;
  int bb = tile / NTILES, ntile = tile % NTILES;
  int n0 = ntile * 16;
  int col_lo = n0 - HALO;                            // slab origin (may be < 0)
  const _Float16* Hb = Ht + (size_t)bb * NCOL * COUT;

#if HAVE_TDM
  if (col_lo >= 0) {
    // fast path: contiguous 1-D copy; elements beyond (NCOL-col_lo)*COUT -> 0
    if (threadIdx.x < 32) {
      tdm_load_2d_f16(0u, (const void*)(Hb + (size_t)col_lo * COUT),
                      (unsigned)(SLABW * COUT), 1u,
                      (unsigned)((NCOL - col_lo) * COUT), 1u,
                      (unsigned)(SLABW * COUT));
      __builtin_amdgcn_s_wait_tensorcnt(0);
    }
  } else {
    // left edge: cooperative copy with explicit zero padding (g < 0)
    for (int idx = threadIdx.x; idx < SLABW * COUT; idx += 256) {
      long gidx = (long)col_lo * COUT + idx;
      slab[idx] = (gidx >= 0) ? Hb[gidx] : (_Float16)0.f;
    }
  }
#else
  for (int idx = threadIdx.x; idx < SLABW * COUT; idx += 256) {
    long gidx = (long)col_lo * COUT + idx;
    slab[idx] = (gidx >= 0 && gidx < (long)NCOL * COUT) ? Hb[gidx] : (_Float16)0.f;
  }
#endif
  __syncthreads();
  lds_written_by_dma(slab);          // slab now holds DMA-written data

  int wave = threadIdx.x >> 5, lane = threadIdx.x & 31;
  int m0 = wave * 16;
  int lm = lane & 15;
  int lh = (lane >> 4) & 1;
  int n = n0 + lm;

  v8f acc = {};
#pragma unroll
  for (int k = 0; k < 9; ++k) {
    int sb = lm + HALO + (k - 4) * V_;               // slab row, in [0, 215]
    const _Float16* Wk = Wt + (size_t)k * COUT * COUT;
#pragma unroll
    for (int c0 = 0; c0 < COUT; c0 += 32) {
      v16h av = load_a16(Wk + (size_t)(m0 + lm) * COUT + c0 + (lh ? 8 : 0));
      v16h bv = load_a16(&slab[sb * COUT + c0 + (lh ? 8 : 0)]);  // 2x ds_load_b128
      acc = __builtin_amdgcn_wmma_f32_16x16x32_f16(false, av, false, bv, (short)0, acc, false, false);
    }
  }
  if (n < NCOL) {
    float* Yb = Y + (size_t)bb * COUT * NCOL;
#pragma unroll
    for (int r = 0; r < 8; ++r) {
      int m = m0 + (lh ? 8 : 0) + r;
      Yb[(size_t)m * NCOL + n] = acc[r] + bias[m];
    }
  }
}

// ---------------------------------------------------------------------------
// BN statistics / finalize / fused elementwise
// ---------------------------------------------------------------------------
__global__ void bn_stats(const float* __restrict__ Y, float* __restrict__ stats) {
  __shared__ float sh[256], shq[256];
  int o = blockIdx.x;
  const size_t per = (size_t)B_ * NCOL;
  float s = 0.f, q = 0.f;
  size_t stride = (size_t)gridDim.y * blockDim.x;
  for (size_t i = (size_t)blockIdx.y * blockDim.x + threadIdx.x; i < per; i += stride) {
    size_t bb = i / NCOL, nn = i % NCOL;
    float v = Y[bb * ((size_t)COUT * NCOL) + (size_t)o * NCOL + nn];
    s += v;
    q += v * v;
  }
  sh[threadIdx.x] = s;
  shq[threadIdx.x] = q;
  __syncthreads();
  for (int w = 128; w > 0; w >>= 1) {
    if ((int)threadIdx.x < w) {
      sh[threadIdx.x]  += sh[threadIdx.x + w];
      shq[threadIdx.x] += shq[threadIdx.x + w];
    }
    __syncthreads();
  }
  if (threadIdx.x == 0) {
    atomicAdd(&stats[o], sh[0]);
    atomicAdd(&stats[COUT + o], shq[0]);
  }
}

// stats layout per group (floats): [0:128)=sum [128:256)=sumsq [256:384)=mean [384:512)=inv
__global__ void bn_finalize(float* __restrict__ stats) {
  int o = threadIdx.x;
  if (o < COUT) {
    const float cnt = (float)((size_t)B_ * NCOL);
    float m = stats[o] / cnt;
    float var = stats[COUT + o] / cnt - m * m;
    stats[2 * COUT + o] = m;
    stats[3 * COUT + o] = rsqrtf(var + BN_EPS);
  }
}

__global__ void final_fuse(const float* __restrict__ H2, const float* __restrict__ RES,
                           const float* __restrict__ st_t, const float* __restrict__ st_r,
                           const float* __restrict__ g_t, const float* __restrict__ be_t,
                           const float* __restrict__ g_r, const float* __restrict__ be_r,
                           float* __restrict__ out) {
  size_t i = (size_t)blockIdx.x * blockDim.x + threadIdx.x;
  if (i >= NY) return;
  int o = (int)((i / NCOL) % COUT);
  float a = (H2[i]  - st_t[2 * COUT + o]) * st_t[3 * COUT + o] * g_t[o] + be_t[o];
  float r = (RES[i] - st_r[2 * COUT + o]) * st_r[3 * COUT + o] * g_r[o] + be_r[o];
  out[i] = fmaxf(a + r, 0.f);
}

// ---------------------------------------------------------------------------
// Launch
// ---------------------------------------------------------------------------
extern "C" void kernel_launch(void* const* d_in, const int* in_sizes, int n_in,
                              void* d_out, int out_size, void* d_ws, size_t ws_size,
                              hipStream_t stream) {
  (void)in_sizes; (void)n_in; (void)out_size; (void)ws_size;
  const float* x     = (const float*)d_in[0];
  const float* A_hat = (const float*)d_in[1];
  const float* w_sp  = (const float*)d_in[2];
  const float* b_sp  = (const float*)d_in[3];
  const float* g_sp  = (const float*)d_in[4];
  const float* be_sp = (const float*)d_in[5];
  const float* w_t   = (const float*)d_in[6];
  const float* b_t   = (const float*)d_in[7];
  const float* g_t   = (const float*)d_in[8];
  const float* be_t  = (const float*)d_in[9];
  const float* w_r   = (const float*)d_in[10];
  const float* b_r   = (const float*)d_in[11];
  const float* g_r   = (const float*)d_in[12];
  const float* be_r  = (const float*)d_in[13];
  float* out = (float*)d_out;

  char* ws = (char*)d_ws;
  _Float16* xh    = (_Float16*)(ws);                          // NX halves
  _Float16* xg    = (_Float16*)(ws + 122880000ull);           // NX halves
  float*    res   = (float*)   (ws + 245760000ull);           // NY f32
  float*    h1    = (float*)   (ws + 368640000ull);           // NY f32
  _Float16* wr16  = (_Float16*)(ws + 491520000ull);           // 8192 halves
  _Float16* wsp16 = (_Float16*)(ws + 491520000ull + 16384);   // 8192 halves
  _Float16* wt16  = (_Float16*)(ws + 491520000ull + 32768);   // 147456 halves
  float*    stats = (float*)   (ws + 491520000ull + 32768 + 294912);
  float* st_r  = stats;
  float* st_sp = stats + 512;
  float* st_t  = stats + 1024;
  _Float16* h1t = xh;             // (B, NCOL, COUT) f16; reuses xh region
  float*    h2  = (float*)xg;     // reuses xg region

  zero_f32<<<(1536 + 255) / 256, 256, 0, stream>>>(stats, 1536);

  cvt_f32_to_f16<<<(8192 + 255) / 256, 256, 0, stream>>>(w_r, wr16, 8192);
  cvt_f32_to_f16<<<(8192 + 255) / 256, 256, 0, stream>>>(w_sp, wsp16, 8192);
  reorder_wt<<<(COUT * COUT * 9 + 255) / 256, 256, 0, stream>>>(w_t, wt16);

  cvt_f32_to_f16<<<(unsigned)((NX + 255) / 256), 256, 0, stream>>>(x, xh, NX);
  graph_mix<<<(unsigned)((NX + 255) / 256), 256, 0, stream>>>(x, A_hat, xg, NX);

  dim3 ggrid(B_ * NTILES);
  gemm_pointwise_f16<<<ggrid, 256, 0, stream>>>(xh, wr16, b_r, res);
  gemm_pointwise_f16<<<ggrid, 256, 0, stream>>>(xg, wsp16, b_sp, h1);

  dim3 sgrid(COUT, 64);
  bn_stats<<<sgrid, 256, 0, stream>>>(res, st_r);
  bn_stats<<<sgrid, 256, 0, stream>>>(h1, st_sp);
  bn_finalize<<<1, 128, 0, stream>>>(st_r);
  bn_finalize<<<1, 128, 0, stream>>>(st_sp);

  // BN + ReLU + transpose -> (B, NCOL, COUT) f16 for the temporal conv
  {
    int tgrid = B_ * ((NCOL + TT - 1) / TT) * (COUT / TT);   // 32*118*2 = 7552
    norm_relu_transpose_f16<<<tgrid, 256, 0, stream>>>(h1, st_sp, g_sp, be_sp, h1t);
  }

  temporal_conv_wmma<<<ggrid, 256, 0, stream>>>(h1t, wt16, b_t, h2);

  bn_stats<<<sgrid, 256, 0, stream>>>(h2, st_t);
  bn_finalize<<<1, 128, 0, stream>>>(st_t);
  final_fuse<<<(unsigned)((NY + 255) / 256), 256, 0, stream>>>(
      h2, res, st_t, st_r, g_t, be_t, g_r, be_r, out);
}
